// Head_292057776685
// MI455X (gfx1250) — compile-verified, matrix operand order checked by reading, not run on hardware
//
#include <hip/hip_runtime.h>
#include <hip/hip_bf16.h>

// Single-head causal attention, B=32, T=2048, C=H=64, fp32 in/out.
// Pass 1: QKV projection (f32 -> f16) via v_wmma_f32_16x16x32_f16.
//         V is written TRANSPOSED (vt[b][h][t]) so pass 2 needs no transpose.
// Pass 2: flash attention (online softmax, base-2 domain) via WMMA,
//         64-key steps, causal mask applied only on the diagonal step.
// Workspace: (q 8MB) + (k 8MB) + (vt 8MB) = 24 MB of f16.

typedef __attribute__((ext_vector_type(16))) _Float16 v16h;
typedef __attribute__((ext_vector_type(8)))  _Float16 v8h;
typedef __attribute__((ext_vector_type(8)))  float    v8f;

#define B_ 32
#define T_ 2048

static __device__ __forceinline__ v16h combine8(v8h lo, v8h hi) {
  v16h r;
#pragma unroll
  for (int i = 0; i < 8; ++i) { r[i] = lo[i]; r[i + 8] = hi[i]; }
  return r;
}

static __device__ __forceinline__ float rowmax16(float v) {
  v = fmaxf(v, __shfl_xor(v, 1));
  v = fmaxf(v, __shfl_xor(v, 2));
  v = fmaxf(v, __shfl_xor(v, 4));
  v = fmaxf(v, __shfl_xor(v, 8));
  return v;
}

static __device__ __forceinline__ float rowsum16(float v) {
  v += __shfl_xor(v, 1);
  v += __shfl_xor(v, 2);
  v += __shfl_xor(v, 4);
  v += __shfl_xor(v, 8);
  return v;
}

// ---------------------------------------------------------------------------
// Pass 1: q/k = x @ W (row-major f16), vt = (x @ Wv)^T per batch.
// Block: 256 thr (8 waves), 128 rows of x.
// ---------------------------------------------------------------------------
__global__ __launch_bounds__(256, 2)
void qkv_proj_kernel(const float* __restrict__ x,
                     const float* __restrict__ Wq,
                     const float* __restrict__ Wk,
                     const float* __restrict__ Wv,
                     _Float16* __restrict__ qh,
                     _Float16* __restrict__ kh,
                     _Float16* __restrict__ vt)
{
  __shared__ __align__(16) _Float16 xs[128][64];      // 16 KB
  __shared__ __align__(16) _Float16 wt[3][64][64];    // 24 KB, W^T: [mat][n][k]

  const int tid = threadIdx.x;
  const int r0  = blockIdx.x * 128;                   // global row (b*T + t)

  const float4* x4 = (const float4*)(x + (size_t)r0 * 64);
#pragma unroll
  for (int i = 0; i < 8; ++i) {
    int c   = tid + i * 256;
    int row = c >> 4, c4 = (c & 15) * 4;
    float4 v = x4[c];
    xs[row][c4 + 0] = (_Float16)v.x;
    xs[row][c4 + 1] = (_Float16)v.y;
    xs[row][c4 + 2] = (_Float16)v.z;
    xs[row][c4 + 3] = (_Float16)v.w;
  }
  for (int i = tid; i < 3 * 64 * 64; i += 256) {
    int mat = i >> 12;
    int rem = i & 4095;               // rem = k*64 + n
    int k   = rem >> 6, n = rem & 63;
    const float* W = (mat == 0) ? Wq : ((mat == 1) ? Wk : Wv);
    wt[mat][n][k] = (_Float16)W[rem];
  }
  __syncthreads();

  const int lane = tid & 31, w = tid >> 5;
  const int hi16 = lane >> 4, lo16 = lane & 15;
  const int kb   = hi16 * 8;
  const int lrow = w * 16 + lo16;
  const int bb   = r0 >> 11;                          // batch index (T = 2048)
  const int t0   = r0 & (T_ - 1);

  v16h a0 = combine8(*(const v8h*)&xs[lrow][kb],      *(const v8h*)&xs[lrow][kb + 16]);
  v16h a1 = combine8(*(const v8h*)&xs[lrow][kb + 32], *(const v8h*)&xs[lrow][kb + 48]);

#pragma unroll
  for (int mat = 0; mat < 3; ++mat) {
#pragma unroll
    for (int nt = 0; nt < 4; ++nt) {
      int n = nt * 16 + lo16;
      v16h b0 = combine8(*(const v8h*)&wt[mat][n][kb],      *(const v8h*)&wt[mat][n][kb + 16]);
      v16h b1 = combine8(*(const v8h*)&wt[mat][n][kb + 32], *(const v8h*)&wt[mat][n][kb + 48]);
      v8f c;
#pragma unroll
      for (int g = 0; g < 8; ++g) c[g] = 0.0f;
      c = __builtin_amdgcn_wmma_f32_16x16x32_f16(false, a0, false, b0, (short)0, c, false, false);
      c = __builtin_amdgcn_wmma_f32_16x16x32_f16(false, a1, false, b1, (short)0, c, false, false);
      if (mat < 2) {
        _Float16* dst = (mat == 0) ? qh : kh;
#pragma unroll
        for (int g = 0; g < 8; ++g) {
          int m = w * 16 + g + 8 * hi16;
          dst[(size_t)(r0 + m) * 64 + n] = (_Float16)c[g];
        }
      } else {
        // transposed V: vt[b][n][t]; consecutive g -> contiguous t
#pragma unroll
        for (int g = 0; g < 8; ++g) {
          int m = w * 16 + g + 8 * hi16;
          vt[(size_t)bb * 64 * T_ + (size_t)n * T_ + (t0 + m)] = (_Float16)c[g];
        }
      }
    }
  }
}

// ---------------------------------------------------------------------------
// Pass 2: causal flash attention. Block: 128 thr (4 wave32s), 64 query rows.
// Key loop steps 64 keys: 8 WMMA for S, 8 WMMA for PV per step.
// ---------------------------------------------------------------------------
__global__ __launch_bounds__(128, 2)
void flash_attn_kernel(const _Float16* __restrict__ qh,
                       const _Float16* __restrict__ kh,
                       const _Float16* __restrict__ vt,
                       float* __restrict__ out)
{
  __shared__ __align__(16) _Float16 Qs[64][64];      // 8 KB
  __shared__ __align__(16) _Float16 Ks[64][64];      // 8 KB (key rows, contiguous c)
  __shared__ __align__(16) _Float16 Vt[64][64];      // 8 KB ([head col][key])
  __shared__ __align__(16) _Float16 Ps[4][16][64];   // 8 KB (per-wave P tile)

  const int tid  = threadIdx.x;
  const int lane = tid & 31, w = tid >> 5;
  const int hi16 = lane >> 4, lo16 = lane & 15;
  const int kb   = hi16 * 8;
  const int b    = blockIdx.x >> 5;                  // 32 query blocks per batch
  const int qblk = blockIdx.x & 31;
  const int qbase = qblk * 64;
  const size_t base   = (size_t)b * T_ * 64;         // q/k base
  const size_t base_v = (size_t)b * 64 * T_;         // vt base

  // Stage Q block once.
  {
    const uint4* src = (const uint4*)(qh + base + (size_t)qbase * 64);
    uint4* dst = (uint4*)&Qs[0][0];
#pragma unroll
    for (int i = 0; i < 4; ++i) dst[tid + i * 128] = src[tid + i * 128];
  }
  __syncthreads();

  const int qrow0 = qbase + w * 16;
  v16h qa0, qa1;
  {
    int lr = w * 16 + lo16;
    qa0 = combine8(*(const v8h*)&Qs[lr][kb],      *(const v8h*)&Qs[lr][kb + 16]);
    qa1 = combine8(*(const v8h*)&Qs[lr][kb + 32], *(const v8h*)&Qs[lr][kb + 48]);
  }

  float m_i[8], l_i[8];
  v8f acc[4];
#pragma unroll
  for (int g = 0; g < 8; ++g) { m_i[g] = -__builtin_huge_valf(); l_i[g] = 0.0f; }
#pragma unroll
  for (int h = 0; h < 4; ++h)
#pragma unroll
    for (int g = 0; g < 8; ++g) acc[h][g] = 0.0f;

  // base-2 softmax domain: 2^(raw * C^-0.5 * log2 e)
  const float scale2 = 0.125f * 1.44269504088896340736f;
  const int nsteps = qblk + 1;                       // 64-key steps

  for (int s = 0; s < nsteps; ++s) {
    const int k0 = s << 6;
    __syncthreads();   // previous-iteration consumers done before overwrite
    {
      // K rows k0..k0+63: straight copy (4096 halves = 512 uint4).
      const uint4* ksrc = (const uint4*)(kh + base + (size_t)k0 * 64);
      uint4* kdst = (uint4*)&Ks[0][0];
#pragma unroll
      for (int i = 0; i < 4; ++i) kdst[tid + i * 128] = ksrc[tid + i * 128];
      // V^T rows h=0..63, cols k0..k0+63 (row stride T in global).
#pragma unroll
      for (int i = 0; i < 4; ++i) {
        int c = tid + i * 128;                       // 512 chunks of 8 halves
        int hrow = c >> 3, c8 = (c & 7) * 8;
        *(uint4*)&Vt[hrow][c8] =
            *(const uint4*)(vt + base_v + (size_t)hrow * T_ + k0 + c8);
      }
      if (s + 1 < nsteps) {                          // global_prefetch_b8
        __builtin_prefetch(kh + base + (size_t)(k0 + 64) * 64 + tid * 32, 0, 0);
        __builtin_prefetch(vt + base_v + (size_t)(tid >> 1) * T_ + (k0 + 64) + (tid & 1) * 32, 0, 0);
      }
    }
    __syncthreads();

    // S = Q (16x64) @ K^T (64x64): four 16-key column tiles, 8 WMMAs.
    v8f sc[4];
#pragma unroll
    for (int jj = 0; jj < 4; ++jj) {
      int nl = jj * 16 + lo16;                       // key row = B-operand col
      v16h b0 = combine8(*(const v8h*)&Ks[nl][kb],      *(const v8h*)&Ks[nl][kb + 16]);
      v16h b1 = combine8(*(const v8h*)&Ks[nl][kb + 32], *(const v8h*)&Ks[nl][kb + 48]);
      v8f sv;
#pragma unroll
      for (int g = 0; g < 8; ++g) sv[g] = 0.0f;
      sv = __builtin_amdgcn_wmma_f32_16x16x32_f16(false, qa0, false, b0, (short)0, sv, false, false);
      sv = __builtin_amdgcn_wmma_f32_16x16x32_f16(false, qa1, false, b1, (short)0, sv, false, false);
      sc[jj] = sv;
    }

    // Scale into log2 domain; causal mask only on the diagonal step.
    if (k0 + 63 > qrow0) {
#pragma unroll
      for (int jj = 0; jj < 4; ++jj)
#pragma unroll
        for (int g = 0; g < 8; ++g) {
          int m = qrow0 + g + 8 * hi16;
          float v = sc[jj][g] * scale2;
          if (k0 + jj * 16 + lo16 > m) v = -__builtin_huge_valf();
          sc[jj][g] = v;
        }
    } else {
#pragma unroll
      for (int jj = 0; jj < 4; ++jj)
#pragma unroll
        for (int g = 0; g < 8; ++g) sc[jj][g] *= scale2;
    }

    // Online softmax (base 2) + P tile to per-wave LDS.
#pragma unroll
    for (int g = 0; g < 8; ++g) {
      float rm = fmaxf(fmaxf(sc[0][g], sc[1][g]), fmaxf(sc[2][g], sc[3][g]));
      rm = rowmax16(rm);
      float mn = fmaxf(m_i[g], rm);
      float alpha = __builtin_amdgcn_exp2f(m_i[g] - mn);   // first step: 0
      float e0 = __builtin_amdgcn_exp2f(sc[0][g] - mn);
      float e1 = __builtin_amdgcn_exp2f(sc[1][g] - mn);
      float e2 = __builtin_amdgcn_exp2f(sc[2][g] - mn);
      float e3 = __builtin_amdgcn_exp2f(sc[3][g] - mn);
      l_i[g] = l_i[g] * alpha + rowsum16((e0 + e1) + (e2 + e3));
      m_i[g] = mn;
#pragma unroll
      for (int h = 0; h < 4; ++h) acc[h][g] *= alpha;
      int m = g + 8 * hi16;
      Ps[w][m][lo16]      = (_Float16)e0;
      Ps[w][m][lo16 + 16] = (_Float16)e1;
      Ps[w][m][lo16 + 32] = (_Float16)e2;
      Ps[w][m][lo16 + 48] = (_Float16)e3;
    }

    // acc (16x64) += P (16x64) @ V (64x64): 8 WMMAs.
    v16h pA0 = combine8(*(const v8h*)&Ps[w][lo16][kb],      *(const v8h*)&Ps[w][lo16][kb + 16]);
    v16h pA1 = combine8(*(const v8h*)&Ps[w][lo16][kb + 32], *(const v8h*)&Ps[w][lo16][kb + 48]);
#pragma unroll
    for (int h = 0; h < 4; ++h) {
      int r = h * 16 + lo16;
      v16h vb0 = combine8(*(const v8h*)&Vt[r][kb],      *(const v8h*)&Vt[r][kb + 16]);
      v16h vb1 = combine8(*(const v8h*)&Vt[r][kb + 32], *(const v8h*)&Vt[r][kb + 48]);
      acc[h] = __builtin_amdgcn_wmma_f32_16x16x32_f16(false, pA0, false, vb0, (short)0, acc[h], false, false);
      acc[h] = __builtin_amdgcn_wmma_f32_16x16x32_f16(false, pA1, false, vb1, (short)0, acc[h], false, false);
    }
  }

  // Epilogue: out = acc / l.
#pragma unroll
  for (int g = 0; g < 8; ++g) {
    float inv = 1.0f / l_i[g];
    int m = qrow0 + g + 8 * hi16;
#pragma unroll
    for (int h = 0; h < 4; ++h)
      out[base + (size_t)m * 64 + h * 16 + lo16] = acc[h][g] * inv;
  }
}

// ---------------------------------------------------------------------------
extern "C" void kernel_launch(void* const* d_in, const int* in_sizes, int n_in,
                              void* d_out, int out_size, void* d_ws, size_t ws_size,
                              hipStream_t stream) {
  (void)in_sizes; (void)n_in; (void)out_size; (void)ws_size;
  const float* x  = (const float*)d_in[0];
  const float* Wq = (const float*)d_in[1];
  const float* Wk = (const float*)d_in[2];
  const float* Wv = (const float*)d_in[3];

  _Float16* qh = (_Float16*)d_ws;                    // 8 MB
  _Float16* kh = qh + (size_t)B_ * T_ * 64;          // 8 MB
  _Float16* vt = kh + (size_t)B_ * T_ * 64;          // 8 MB (transposed V)
  float* out = (float*)d_out;

  qkv_proj_kernel<<<(B_ * T_) / 128, 256, 0, stream>>>(x, Wq, Wk, Wv, qh, kh, vt);
  flash_attn_kernel<<<B_ * (T_ / 64), 128, 0, stream>>>(qh, kh, vt, out);
}
